// NeuralODE_90546500534549
// MI455X (gfx1250) — matrix operand order repaired and checked by reading, not compiled, not run on hardware
//
#include <hip/hip_runtime.h>
#include <hip/hip_bf16.h>

// CDNA5 / gfx1250 wave32 WMMA types
typedef __attribute__((ext_vector_type(16))) _Float16 v16h;
typedef __attribute__((ext_vector_type(8)))  float    v8f;

#define WREG 134  // LDS row stride (halfs) for W2^T = 67 dwords -> bank-conflict-free B loads

union HFrag {
  v16h     v;
  _Float16 e[16];
  unsigned u[8];
};

// A-fragment K pattern (16-bit A 16x32, per CDNA5 ISA 7.12.2):
// VGPR p (lo lanes): p<4 -> K=2p,2p+1 ; p>=4 -> K=16+2(p-4) ; hi lanes +8
#define APAT(p) (((p) < 4) ? (2 * (p)) : (16 + 2 * ((p) - 4)))

__global__ __launch_bounds__(256) void NeuralODE_siren_rk4_kernel(
    const float* __restrict__ gt,  const float* __restrict__ gx0,
    const float* __restrict__ gW1, const float* __restrict__ gb1,
    const float* __restrict__ gW2, const float* __restrict__ gb2,
    const float* __restrict__ gW3, const float* __restrict__ gb3,
    float* __restrict__ gout, int N, int T)
{
  __shared__ _Float16 sW2T[128][WREG];           // W2 transposed, f16
  __shared__ float sW1[256];                     // [2][128]
  __shared__ float sB1[128], sB2[128];
  __shared__ float sW3[128][2];
  __shared__ float sB3[2];
  __shared__ float sT[128];
  __shared__ float sY[128][2],  sYt[128][2];
  __shared__ float sK1[128][2], sK2[128][2], sK3[128][2], sK4[128][2];

  const int tid     = threadIdx.x;
  const int lane    = tid & 31;          // wave32
  const int wave    = tid >> 5;          // 8 waves / block
  const int rowbase = wave * 16;         // 16 trajectories per wave
  const int c0      = lane & 15;
  const int hh      = lane >> 4;         // half-wave select
  const size_t nbase2  = (size_t)blockIdx.x * 256;  // block's flat float offset
  const size_t stride2 = (size_t)N * 2;
  const float  w0 = 44.0f;

  // ---- one-time cooperative weight staging (L2-resident, trivial cost) ----
  for (int i = tid; i < 128 * 128; i += 256)
    sW2T[i & 127][i >> 7] = (_Float16)gW2[i];     // W2T[n][k] = W2[k][n]
  sW1[tid] = gW1[tid];                             // 2*128 == 256
  if (tid < 128) { sB1[tid] = gb1[tid]; sB2[tid] = gb2[tid]; }
  sW3[tid >> 1][tid & 1] = gW3[tid];               // 128*2 == 256
  if (tid < 2)  sB3[tid] = gb3[tid];
  if (tid < T)  sT[tid]  = gt[tid];
  {
    float v = gx0[nbase2 + tid];
    sY[tid >> 1][tid & 1] = v;
    gout[nbase2 + tid] = v;                        // trajectory slab t=0
  }
  __syncthreads();

  // ---- one SIREN vector-field evaluation for this wave's 16 points ----
  auto feval = [&](const float (*in)[2], float (*out)[2]) {
    // layer 1 (K=2) computed per-lane straight into the WMMA A-fragment layout
    const int m = rowbase + c0;
    const float x0v = in[m][0], x1v = in[m][1];
    HFrag a[4];
#pragma unroll
    for (int kt = 0; kt < 4; ++kt) {
#pragma unroll
      for (int p = 0; p < 8; ++p) {
        const int kb = 32 * kt + APAT(p) + hh * 8;
#pragma unroll
        for (int e2 = 0; e2 < 2; ++e2) {
          const int k = kb + e2;
          float pre = fmaf(x1v, sW1[128 + k], fmaf(x0v, sW1[k], sB1[k]));
          a[kt].e[2 * p + e2] = (_Float16)__sinf(w0 * pre);
        }
      }
    }

    // layer 2: 16x128 += (16x128)(128x128) as 4 K-tiles x 8 N-tiles of WMMA
    v8f c[8] = {};
#pragma unroll
    for (int kt = 0; kt < 4; ++kt) {
#pragma unroll
      for (int nt = 0; nt < 8; ++nt) {
        HFrag b;
        const int n = c0 + 16 * nt;                 // B column (lo & hi halves pair up)
#pragma unroll
        for (int p = 0; p < 8; ++p) {
          const int kb = 32 * kt + APAT(p) + hh * 8; // B: same K pattern as A
          b.u[p] = *reinterpret_cast<const unsigned*>(&sW2T[n][kb]);
        }
        c[nt] = __builtin_amdgcn_wmma_f32_16x16x32_f16(
            false, a[kt].v, false, b.v, (short)0, c[nt], false, false);
      }
    }

    // layer 3 (N=2) epilogue: per-lane partials, butterfly reduce over 16 lanes
    float acc[8][2];
#pragma unroll
    for (int j = 0; j < 8; ++j) { acc[j][0] = 0.f; acc[j][1] = 0.f; }
#pragma unroll
    for (int nt = 0; nt < 8; ++nt) {
      const int n = c0 + 16 * nt;
      const float w30 = sW3[n][0], w31 = sW3[n][1], bb = sB2[n];
#pragma unroll
      for (int j = 0; j < 8; ++j) {
        float h2 = __sinf(w0 * (c[nt][j] + bb));
        acc[j][0] = fmaf(h2, w30, acc[j][0]);
        acc[j][1] = fmaf(h2, w31, acc[j][1]);
      }
    }
#pragma unroll
    for (int mk = 1; mk <= 8; mk <<= 1) {
#pragma unroll
      for (int j = 0; j < 8; ++j) {
        acc[j][0] += __shfl_xor(acc[j][0], mk, 32);
        acc[j][1] += __shfl_xor(acc[j][1], mk, 32);
      }
    }
    if (c0 == 0) {                                  // lane0 -> rows 0-7, lane16 -> rows 8-15
      const int rb = rowbase + (hh ? 8 : 0);
#pragma unroll
      for (int j = 0; j < 8; ++j) {
        out[rb + j][0] = acc[j][0] + sB3[0];
        out[rb + j][1] = acc[j][1] + sB3[1];
      }
    }
    __builtin_amdgcn_wave_barrier();
  };

  // ---- RK4 (3/8 rule) time loop; all cross-lane traffic is wave-private LDS ----
  const int mm = rowbase + (lane >> 1);
  const int dd = lane & 1;
  const float one3 = 1.0f / 3.0f;

  for (int s = 0; s < T - 1; ++s) {
    const float dt = sT[s + 1] - sT[s];

    feval(sY, sK1);
    sYt[mm][dd] = sY[mm][dd] + dt * sK1[mm][dd] * one3;
    __builtin_amdgcn_wave_barrier();

    feval(sYt, sK2);
    sYt[mm][dd] = sY[mm][dd] + dt * (sK2[mm][dd] - sK1[mm][dd] * one3);
    __builtin_amdgcn_wave_barrier();

    feval(sYt, sK3);
    sYt[mm][dd] = sY[mm][dd] + dt * (sK1[mm][dd] - sK2[mm][dd] + sK3[mm][dd]);
    __builtin_amdgcn_wave_barrier();

    feval(sYt, sK4);
    {
      float yn = sY[mm][dd] +
                 dt * 0.125f * (sK1[mm][dd] + 3.0f * (sK2[mm][dd] + sK3[mm][dd]) + sK4[mm][dd]);
      sY[mm][dd] = yn;
      gout[(size_t)(s + 1) * stride2 + nbase2 + tid] = yn;  // coalesced: flat idx == tid
    }
    __builtin_amdgcn_wave_barrier();
  }
}

extern "C" void kernel_launch(void* const* d_in, const int* in_sizes, int n_in,
                              void* d_out, int out_size, void* d_ws, size_t ws_size,
                              hipStream_t stream) {
  const float* t  = (const float*)d_in[0];
  const float* x0 = (const float*)d_in[1];
  const float* W1 = (const float*)d_in[2];
  const float* b1 = (const float*)d_in[3];
  const float* W2 = (const float*)d_in[4];
  const float* b2 = (const float*)d_in[5];
  const float* W3 = (const float*)d_in[6];
  const float* b3 = (const float*)d_in[7];
  float* out = (float*)d_out;

  const int N = in_sizes[1] / 2;   // 262144
  const int T = in_sizes[0];       // 101
  const int blocks = N / 128;      // 128 points per block (8 waves x 16)

  hipLaunchKernelGGL(NeuralODE_siren_rk4_kernel, dim3(blocks), dim3(256), 0, stream,
                     t, x0, W1, b1, W2, b2, W3, b3, out, N, T);
}
